// RVTDCNN_42760694399004
// MI455X (gfx1250) — matrix operand order repaired and checked by reading, not compiled, not run on hardware
//
#include <hip/hip_runtime.h>
#include <hip/hip_bf16.h>

typedef __attribute__((ext_vector_type(16))) _Float16 v16h;
typedef __attribute__((ext_vector_type(8)))  _Float16 v8h;
typedef __attribute__((ext_vector_type(8)))  float    v8f;

#define WAVES   8
#define TILE_M  16
#define HF      16      // feature dim H
#define FC_IN   96      // 3 * 2 * 16
#define NROWS   19      // 16 samples + 3 halo rows

__device__ __forceinline__ float fast_tanh(float x) {
    // tanh(x) = 1 - 2/(e^{2x}+1), e^{2x} = exp2(x * 2*log2(e))
    float t = __builtin_amdgcn_exp2f(x * 2.885390081777927f);
    return 1.0f - 2.0f * __builtin_amdgcn_rcpf(t + 1.0f);
}

// Butterfly sum across the 16 lanes of a half-wave (xor 1,2,4,8 stays in-half).
__device__ __forceinline__ float half_reduce(float v) {
    v += __int_as_float(__builtin_amdgcn_ds_swizzle(__float_as_int(v), 0x041f)); // xor 1
    v += __int_as_float(__builtin_amdgcn_ds_swizzle(__float_as_int(v), 0x081f)); // xor 2
    v += __int_as_float(__builtin_amdgcn_ds_swizzle(__float_as_int(v), 0x101f)); // xor 4
    v += __int_as_float(__builtin_amdgcn_ds_swizzle(__float_as_int(v), 0x201f)); // xor 8
    return v;
}

__global__ __launch_bounds__(256) void rvtdcnn_fused_kernel(
    const float* __restrict__ x,        // (B*T, 16)
    const float* __restrict__ conv_w,   // (3,1,3,3) = 27
    const float* __restrict__ conv_b,   // 3
    const float* __restrict__ fc1_w,    // (6, 96)
    const float* __restrict__ fc1_b,    // 6
    const float* __restrict__ fc2_w,    // (2, 6)
    const float* __restrict__ fc2_b,    // 2
    float* __restrict__ out,            // (B*T, 2)
    int T, int ntiles)
{
    __shared__ __align__(16) float    xs[WAVES][NROWS * HF];      // staged input rows
    __shared__ __align__(16) _Float16 act[WAVES][TILE_M * FC_IN]; // conv activations (f16)

    const int wid   = threadIdx.x >> 5;
    const int lane  = threadIdx.x & 31;
    const int nlane = lane & 15;   // column within half-wave
    const int half  = lane >> 4;   // 0/1 -> conv output position w, WMMA half

    int tile = blockIdx.x * WAVES + wid;
    if (tile >= ntiles) tile = ntiles - 1;     // wave-uniform clamp (benign dup work)
    const int s0 = tile * TILE_M;              // first flat sample (b*T + t)
    const int t0 = s0 % T;                     // t within batch row

    // ---- uniform weights ----
    float cw[27];
#pragma unroll
    for (int i = 0; i < 27; ++i) cw[i] = conv_w[i];
    const float cb[3] = { conv_b[0], conv_b[1], conv_b[2] };

    const float hb  = (nlane < 6) ? fc1_b[nlane]     : 0.0f;
    const float w20 = (nlane < 6) ? fc2_w[nlane]     : 0.0f;  // fc2_w[0][n]
    const float w21 = (nlane < 6) ? fc2_w[6 + nlane] : 0.0f;  // fc2_w[1][n]
    const float bo0 = fc2_b[0], bo1 = fc2_b[1];

    // ---- B fragments: B[k][n] = fc1_w[n][k], N padded 6->16, K = 3 tiles of 32 ----
    // Dense f16 B 32x16 layout: lanes 0-15 hold K{0..7,16..23} (N=lane), lanes 16-31 K{8..15,24..31}.
    v16h bfrag[3];
#pragma unroll
    for (int kt = 0; kt < 3; ++kt) {
#pragma unroll
        for (int i = 0; i < 16; ++i) {
            int k = kt * 32 + i + (i >= 8 ? 8 : 0) + (half ? 8 : 0);
            bfrag[kt][i] = (nlane < 6) ? (_Float16)fc1_w[nlane * FC_IN + k]
                                       : (_Float16)0.0f;
        }
    }

    // ---- stage input rows (s0-3 .. s0+15), contiguous, 64B-aligned float4 loads ----
    {
        const float4* src = (const float4*)(x + (long)(s0 - 3) * HF);
        for (int e4 = lane; e4 < (NROWS * HF) / 4; e4 += 32) {
            const int r = e4 >> 2;                    // row index 0..18
            float4 v = make_float4(0.f, 0.f, 0.f, 0.f);
            if (!(t0 == 0 && r < 3)) v = src[e4];     // zero-pad t<0 (never deref OOB)
            *(float4*)&xs[wid][e4 * 4] = v;
        }
    }
    __syncthreads();

    // ---- conv 3x3 (H-padded) + tanh -> f16 activations in FC-input order ----
    {
        const int j = nlane;
        // Branch-free halo: unconditional loads from clamped addresses,
        // zeroed with v_cndmask (no exec-mask branch ladders).
        const int   jm1   = (j > 0)  ? j - 1 : j;
        const int   jp1   = (j < 15) ? j + 1 : j;
        const bool  has_l = (j > 0);
        const bool  has_r = (j < 15);
        const float* rows = &xs[wid][0];
        float L0, M0, R0, L1, M1, R1, L2, M2, R2;

        auto LD = [&](int r, float& L, float& M, float& R) {
            const float* p = rows + r * HF;
            float l = p[jm1];               // always-valid address
            M = p[j];
            float rr = p[jp1];              // always-valid address
            L = has_l ? l  : 0.0f;          // v_cndmask
            R = has_r ? rr : 0.0f;          // v_cndmask
        };

        LD(half,     L0, M0, R0);
        LD(half + 1, L1, M1, R1);
#pragma unroll 4
        for (int m = 0; m < TILE_M; ++m) {
            LD(m + half + 2, L2, M2, R2);
#pragma unroll
            for (int c = 0; c < 3; ++c) {
                const float* k9 = &cw[c * 9];
                float a = cb[c];
                a = fmaf(L0, k9[0], a); a = fmaf(M0, k9[1], a); a = fmaf(R0, k9[2], a);
                a = fmaf(L1, k9[3], a); a = fmaf(M1, k9[4], a); a = fmaf(R1, k9[5], a);
                a = fmaf(L2, k9[6], a); a = fmaf(M2, k9[7], a); a = fmaf(R2, k9[8], a);
                // k = c*32 + w*16 + j  (matches reference reshape order)
                act[wid][m * FC_IN + c * 32 + half * 16 + j] = (_Float16)fast_tanh(a);
            }
            L0 = L1; M0 = M1; R0 = R1;
            L1 = L2; M1 = M2; R1 = R2;
        }
    }
    __syncthreads();

    // ---- FC1 via WMMA: (16 x 96) @ (96 x 16pad), 3 chained 16x16x32 f16 MMAs ----
    v8f acc = {0.f, 0.f, 0.f, 0.f, 0.f, 0.f, 0.f, 0.f};
    {
        const _Float16* ab = &act[wid][nlane * FC_IN];  // lane = M row
#pragma unroll
        for (int kt = 0; kt < 3; ++kt) {
            // A 16x32 f16 layout: lanes 0-15 K{0..7,16..23}, lanes 16-31 K{8..15,24..31}
            const _Float16* p = ab + kt * 32 + (half ? 8 : 0);
            v8h lo = *(const v8h*)(p);        // K base+0..7   (ds_load_b128)
            v8h hi = *(const v8h*)(p + 16);   // K base+16..23 (ds_load_b128)
            v16h a = __builtin_shufflevector(lo, hi,
                0, 1, 2, 3, 4, 5, 6, 7, 8, 9, 10, 11, 12, 13, 14, 15);
            acc = __builtin_amdgcn_wmma_f32_16x16x32_f16(
                false, a, false, bfrag[kt], (short)0, acc, false, false);
        }
    }

    // ---- bias + tanh + FC2 (6->2) via half-wave butterfly, write 2 outputs ----
    float y0[8], y1[8];
#pragma unroll
    for (int r = 0; r < 8; ++r) {
        float h  = fast_tanh(acc[r] + hb);   // n>=6 columns: h = tanh(0) with w2=0
        float p0 = half_reduce(h * w20);
        float p1 = half_reduce(h * w21);
        y0[r] = p0 + bo0;
        y1[r] = p1 + bo1;
    }
    if (nlane == 0) {
        // C layout: lanes<16 hold M=0..7, lanes>=16 hold M=8..15 (VGPR r = row)
        float* op = out + (long)(s0 + half * 8) * 2;
#pragma unroll
        for (int g = 0; g < 4; ++g) {
            float4 v = make_float4(y0[2 * g], y1[2 * g], y0[2 * g + 1], y1[2 * g + 1]);
            *(float4*)(op + 4 * g) = v;
        }
    }
}

extern "C" void kernel_launch(void* const* d_in, const int* in_sizes, int n_in,
                              void* d_out, int out_size, void* d_ws, size_t ws_size,
                              hipStream_t stream) {
    const float* x      = (const float*)d_in[0];
    const float* conv_w = (const float*)d_in[1];
    const float* conv_b = (const float*)d_in[2];
    const float* fc1_w  = (const float*)d_in[3];
    const float* fc1_b  = (const float*)d_in[4];
    const float* fc2_w  = (const float*)d_in[5];
    const float* fc2_b  = (const float*)d_in[6];
    float* out = (float*)d_out;

    const int T = 16384;                       // per reference
    const int n = in_sizes[0] / HF;            // B*T samples
    const int ntiles = (n + TILE_M - 1) / TILE_M;
    const int blocks = (ntiles + WAVES - 1) / WAVES;

    rvtdcnn_fused_kernel<<<blocks, 256, 0, stream>>>(
        x, conv_w, conv_b, fc1_w, fc1_b, fc2_w, fc2_b, out, T, ntiles);
}
